// VisionMambaModel_14156212207682
// MI455X (gfx1250) — matrix operand order repaired and checked by reading, not compiled, not run on hardware
//
#include <hip/hip_runtime.h>
#include <math.h>

typedef __attribute__((ext_vector_type(16))) _Float16 v16h;
typedef __attribute__((ext_vector_type(8)))  _Float16 v8h;
typedef __attribute__((ext_vector_type(8)))  float    v8f;

#define B_SZ     8
#define L_TOK    1025
#define R_ROWS   (B_SZ * L_TOK)      // 8200
#define DM       128
#define DI       256
#define DS       16
#define DTR      8
#define NPATCH   1024                // 32x32 patches per image
#define PROWS    (B_SZ * NPATCH)     // 8192

static __device__ __forceinline__ float siluf(float x) {
    return x / (1.0f + __expf(-x));
}
static __device__ __forceinline__ float geluf(float x) {
    return 0.5f * x * (1.0f + erff(x * 0.70710678118654752f));
}

// ---------------------------------------------------------------------------
// f32 -> f16 conversion (used once per launch for weights)
// ---------------------------------------------------------------------------
__global__ void f32_to_f16_kernel(const float* __restrict__ src,
                                  _Float16* __restrict__ dst, int n)
{
    int i = blockIdx.x * blockDim.x + threadIdx.x;
    if (i < n) dst[i] = (_Float16)src[i];
}

// ---------------------------------------------------------------------------
// WMMA GEMM: out[M,N] = act( A[M,K](f16) * W[N,K]^T + bias + res )
// One wave computes a 16x64 strip: one A fragment reused across 4 B fragments
// -> 4 v_wmma_f32_16x16x32_f16 per K step. Fragments are loaded as aligned
// v8h vectors (fp16 in memory; no cvt chain). Register layouts per CDNA5 ISA
// 7.12.2. act: 0 = none, 1 = exact GELU. out (f32) and outh (f16) optional.
// ---------------------------------------------------------------------------
__global__ void gemm_wmma_h(const _Float16* __restrict__ A, int lda,
                            const _Float16* __restrict__ W, int ldw,
                            const float* __restrict__ bias,
                            const float* __restrict__ res, int ldr,
                            float* __restrict__ out, int ldo,
                            _Float16* __restrict__ outh, int ldoh,
                            int M, int N, int K, int act)
{
    const int row0 = blockIdx.x * 16;
    const int n0   = blockIdx.y * 64;
    const int lane = threadIdx.x;          // 0..31, wave32
    const int m16  = lane & 15;
    const int half = lane >> 4;

    int arow = row0 + m16; if (arow >= M) arow = M - 1;   // clamp (writes guarded)
    const _Float16* Arow = A + (size_t)arow * lda;

    const _Float16* Wrow[4];
#pragma unroll
    for (int tn = 0; tn < 4; ++tn) {
        int bn  = n0 + tn * 16 + m16;
        int bnc = (bn < N) ? bn : (N - 1);
        Wrow[tn] = W + (size_t)bnc * ldw;
    }

    v8f acc[4];
#pragma unroll
    for (int tn = 0; tn < 4; ++tn)
        acc[tn] = (v8f){0.f, 0.f, 0.f, 0.f, 0.f, 0.f, 0.f, 0.f};

    for (int kk = 0; kk < K; kk += 32) {
        if (kk + 32 < K) {
            // CDNA5 global_prefetch: pull next K tile toward the WGP
            __builtin_prefetch(Arow + kk + 32, 0, 1);
            __builtin_prefetch(Wrow[0] + kk + 32, 0, 1);
            __builtin_prefetch(Wrow[2] + kk + 32, 0, 1);
        }
        // A 16x32 f16: lane half 0 holds K kk+0..7 / kk+16..23 of its row,
        // half 1 holds K kk+8..15 / kk+24..31 (ISA 7.12.2, 16-bit A layout).
        const v8h* ap  = (const v8h*)(Arow + kk + 8 * half);
        v8h alo = ap[0];
        v8h ahi = ap[2];     // +16 halfs
        v16h a;
#pragma unroll
        for (int i = 0; i < 8; ++i) { a[i] = alo[i]; a[8 + i] = ahi[i]; }

#pragma unroll
        for (int tn = 0; tn < 4; ++tn) {
            // B 32x16 f16: lanes 0-15 hold K kk+0..7 / kk+16..23 of column n; 16-31 K kk+8..15 / kk+24..31.
            const v8h* bp = (const v8h*)(Wrow[tn] + kk + 8 * half);
            v8h b0 = bp[0];
            v8h b1 = bp[2];
            v16h b;
#pragma unroll
            for (int i = 0; i < 8; ++i) { b[i] = b0[i]; b[8 + i] = b1[i]; }
            acc[tn] = __builtin_amdgcn_wmma_f32_16x16x32_f16(false, a, false, b,
                                                             (short)0, acc[tn],
                                                             false, false);
        }
    }

#pragma unroll
    for (int tn = 0; tn < 4; ++tn) {
        const int on = n0 + tn * 16 + m16;
        if (on < N) {
            const float bv = bias ? bias[on] : 0.0f;
#pragma unroll
            for (int r = 0; r < 8; ++r) {
                // D 16x16 f32: VGPR r -> row (r + 8*half), col = lane&15
                const int orow = row0 + r + 8 * half;
                if (orow < M) {
                    float v = acc[tn][r] + bv;
                    if (res) v += res[(size_t)orow * ldr + on];
                    if (act == 1) v = geluf(v);
                    if (out)  out[(size_t)orow * ldo + on] = v;
                    if (outh) outh[(size_t)orow * ldoh + on] = (_Float16)v;
                }
            }
        }
    }
}

// ---------------------------------------------------------------------------
// im2col: (8,1,512,512) -> patches_h (8192, 256) fp16; patch k = ky*16+kx
// ---------------------------------------------------------------------------
__global__ void im2col_kernel(const float* __restrict__ img, _Float16* __restrict__ patches)
{
    int idx = blockIdx.x * blockDim.x + threadIdx.x;
    if (idx >= PROWS * 256) return;
    int k = idx & 255;
    int r = idx >> 8;
    int b = r >> 10;
    int p = r & 1023;
    int py = p >> 5, px = p & 31;
    int ky = k >> 4, kx = k & 15;
    patches[idx] = (_Float16)img[((size_t)b * 512 + (py * 16 + ky)) * 512 + (px * 16 + kx)];
}

// ---------------------------------------------------------------------------
// Assemble token sequence: cls + patch embeddings + patch_b + pos_embed
// ---------------------------------------------------------------------------
__global__ void assemble_x_kernel(const float* __restrict__ xpatch,
                                  const float* __restrict__ patch_b,
                                  const float* __restrict__ cls_tok,
                                  const float* __restrict__ pos,
                                  float* __restrict__ x)
{
    int idx = blockIdx.x * blockDim.x + threadIdx.x;
    if (idx >= R_ROWS * DM) return;
    int n  = idx & (DM - 1);
    int bl = idx >> 7;
    int b  = bl / L_TOK;
    int l  = bl % L_TOK;
    float v;
    if (l == 0) v = cls_tok[n];
    else        v = xpatch[((size_t)(b * NPATCH + l - 1)) * DM + n] + patch_b[n];
    x[(size_t)bl * DM + n] = v + pos[l * DM + n];
}

// ---------------------------------------------------------------------------
// LayerNorm over last dim (128): one row per 128-thread block, fp16 output
// (feeds WMMA GEMMs only)
// ---------------------------------------------------------------------------
__global__ void layernorm_kernel(const float* __restrict__ x,
                                 const float* __restrict__ w,
                                 const float* __restrict__ b,
                                 _Float16* __restrict__ out)
{
    __shared__ float red[DM];
    const int row = blockIdx.x;
    const int t   = threadIdx.x;
    float v = x[(size_t)row * DM + t];
    red[t] = v; __syncthreads();
    for (int s = 64; s > 0; s >>= 1) { if (t < s) red[t] += red[t + s]; __syncthreads(); }
    float mean = red[0] * (1.0f / DM);
    __syncthreads();
    float d = v - mean;
    red[t] = d * d; __syncthreads();
    for (int s = 64; s > 0; s >>= 1) { if (t < s) red[t] += red[t + s]; __syncthreads(); }
    float inv = rsqrtf(red[0] * (1.0f / DM) + 1e-5f);
    out[(size_t)row * DM + t] = (_Float16)(d * inv * w[t] + b[t]);
}

// ---------------------------------------------------------------------------
// Causal depthwise conv (width 4) + SiLU. rev=1 runs the reversed-sequence
// conv expressed in original token order: out[l] = sum_j w[j]*u[l+3-j].
// Writes f32 (scan input) and f16 (x_proj GEMM input).
// ---------------------------------------------------------------------------
__global__ void conv_silu_kernel(const float* __restrict__ xz,
                                 const float* __restrict__ convw,
                                 const float* __restrict__ convb,
                                 float* __restrict__ u,
                                 _Float16* __restrict__ uh, int rev)
{
    int idx = blockIdx.x * blockDim.x + threadIdx.x;
    if (idx >= R_ROWS * DI) return;
    int d  = idx & (DI - 1);
    int bl = idx >> 8;
    int b  = bl / L_TOK;
    int l  = bl % L_TOK;
    const float* w = convw + d * 4;
    float acc = convb[d];
    if (!rev) {
#pragma unroll
        for (int j = 0; j < 4; ++j) {
            int ls = l - 3 + j;
            if (ls >= 0) acc += w[j] * xz[((size_t)(b * L_TOK + ls)) * 512 + d];
        }
    } else {
#pragma unroll
        for (int j = 0; j < 4; ++j) {
            int ls = l + 3 - j;
            if (ls < L_TOK) acc += w[j] * xz[((size_t)(b * L_TOK + ls)) * 512 + d];
        }
    }
    float s = siluf(acc);
    u[(size_t)bl * DI + d]  = s;
    uh[(size_t)bl * DI + d] = (_Float16)s;
}

// ---------------------------------------------------------------------------
// dt = softplus(xdbl[:, :8] @ Wdt^T + bdt)   (K=8: plain VALU)
// ---------------------------------------------------------------------------
__global__ void dt_kernel(const float* __restrict__ xdbl, int ldx,
                          const float* __restrict__ Wdt,
                          const float* __restrict__ bdt,
                          float* __restrict__ dt)
{
    int idx = blockIdx.x * blockDim.x + threadIdx.x;
    if (idx >= R_ROWS * DI) return;
    int d = idx & (DI - 1);
    int r = idx >> 8;
    float acc = bdt[d];
    const float* xr = xdbl + (size_t)r * ldx;
    const float* wr = Wdt + d * DTR;
#pragma unroll
    for (int j = 0; j < DTR; ++j) acc += xr[j] * wr[j];
    dt[(size_t)r * DI + d] = (acc > 20.0f) ? acc : log1pf(__expf(acc));
}

// ---------------------------------------------------------------------------
// Selective scan: one block per batch, 256 threads = 256 channels, state
// h[16] in registers, B/C vectors staged via LDS each step. Gate with
// silu(z) fused. rev=1 scans tokens back-to-front. fp16 output (GEMM input).
// ---------------------------------------------------------------------------
__global__ void scan_kernel(const float* __restrict__ dtb,
                            const float* __restrict__ ub,
                            const float* __restrict__ xdbl, int ldx,
                            const float* __restrict__ xz,
                            const float* __restrict__ Alog,
                            const float* __restrict__ Dp,
                            _Float16* __restrict__ y, int rev)
{
    __shared__ float Bs[DS], Cs[DS];
    const int b = blockIdx.x;
    const int d = threadIdx.x;

    float Ad[DS];
#pragma unroll
    for (int s = 0; s < DS; ++s) Ad[s] = -__expf(Alog[d * DS + s]);
    float h[DS];
#pragma unroll
    for (int s = 0; s < DS; ++s) h[s] = 0.0f;
    const float Dd = Dp[d];

    for (int t = 0; t < L_TOK; ++t) {
        const int l = rev ? (L_TOK - 1 - t) : t;
        const size_t row = (size_t)(b * L_TOK + l);
        __syncthreads();
        if (d < DS)            Bs[d]      = xdbl[row * ldx + DTR + d];
        else if (d < 2 * DS)   Cs[d - DS] = xdbl[row * ldx + DTR + DS + (d - DS)];
        __syncthreads();

        const float dtv = dtb[row * DI + d];
        const float uv  = ub[row * DI + d];
        const float du  = dtv * uv;
        float yv = 0.0f;
#pragma unroll
        for (int s = 0; s < DS; ++s) {
            float dA = __expf(dtv * Ad[s]);
            h[s] = dA * h[s] + du * Bs[s];
            yv  += h[s] * Cs[s];
        }
        yv += uv * Dd;
        const float z = xz[row * 512 + DI + d];
        y[row * DI + d] = (_Float16)(yv * siluf(z));
    }
}

// ---------------------------------------------------------------------------
// Final head: LN(cls token) -> GELU(128->32) -> (32->1)
// ---------------------------------------------------------------------------
__global__ void head_kernel(const float* __restrict__ x,
                            const float* __restrict__ nfw, const float* __restrict__ nfb,
                            const float* __restrict__ h1w, const float* __restrict__ h1b,
                            const float* __restrict__ h2w, const float* __restrict__ h2b,
                            float* __restrict__ out)
{
    __shared__ float red[DM];
    __shared__ float xn[DM];
    __shared__ float hh[32];
    const int b = blockIdx.x;
    const int t = threadIdx.x;
    float v = x[(size_t)(b * L_TOK) * DM + t];
    red[t] = v; __syncthreads();
    for (int s = 64; s > 0; s >>= 1) { if (t < s) red[t] += red[t + s]; __syncthreads(); }
    float mean = red[0] * (1.0f / DM);
    __syncthreads();
    float d = v - mean;
    red[t] = d * d; __syncthreads();
    for (int s = 64; s > 0; s >>= 1) { if (t < s) red[t] += red[t + s]; __syncthreads(); }
    float inv = rsqrtf(red[0] * (1.0f / DM) + 1e-5f);
    xn[t] = d * inv * nfw[t] + nfb[t];
    __syncthreads();
    if (t < 32) {
        float a = h1b[t];
        const float* wr = h1w + t * DM;
        for (int j = 0; j < DM; ++j) a += xn[j] * wr[j];
        hh[t] = geluf(a);
    }
    __syncthreads();
    if (t == 0) {
        float a = h2b[0];
        for (int j = 0; j < 32; ++j) a += hh[j] * h2w[j];
        out[b] = a;
    }
}

// ---------------------------------------------------------------------------
// Host orchestration
// ---------------------------------------------------------------------------
static inline void launch_gemm(hipStream_t s, const _Float16* A, int lda,
                               const _Float16* W, int ldw, const float* bias,
                               const float* res, int ldr,
                               float* out, int ldo,
                               _Float16* outh, int ldoh,
                               int M, int N, int K, int act)
{
    dim3 grid((M + 15) / 16, (N + 63) / 64);
    gemm_wmma_h<<<grid, 32, 0, s>>>(A, lda, W, ldw, bias, res, ldr,
                                    out, ldo, outh, ldoh, M, N, K, act);
}

static inline void launch_cvt(hipStream_t s, const float* src, _Float16* dst, int n)
{
    f32_to_f16_kernel<<<(n + 255) / 256, 256, 0, s>>>(src, dst, n);
}

extern "C" void kernel_launch(void* const* d_in, const int* in_sizes, int n_in,
                              void* d_out, int out_size, void* d_ws, size_t ws_size,
                              hipStream_t stream)
{
    const float* img     = (const float*)d_in[0];
    const float* patch_w = (const float*)d_in[1];   // (128, 256) flattened
    const float* patch_b = (const float*)d_in[2];
    const float* cls_tok = (const float*)d_in[3];
    const float* pos     = (const float*)d_in[4];
    const float* n1w     = (const float*)d_in[5];
    const float* n1b     = (const float*)d_in[6];
    const float* mWin    = (const float*)d_in[7];   // (6,2,512,128)
    const float* mConvw  = (const float*)d_in[8];   // (6,2,256,4)
    const float* mConvb  = (const float*)d_in[9];   // (6,2,256)
    const float* mWx     = (const float*)d_in[10];  // (6,2,40,256)
    const float* mWdt    = (const float*)d_in[11];  // (6,2,256,8)
    const float* mbdt    = (const float*)d_in[12];  // (6,2,256)
    const float* mAlog   = (const float*)d_in[13];  // (6,2,256,16)
    const float* mDp     = (const float*)d_in[14];  // (6,2,256)
    const float* mWout   = (const float*)d_in[15];  // (6,2,128,256)
    const float* mergew  = (const float*)d_in[16];  // (6,128,256)
    const float* mergeb  = (const float*)d_in[17];  // (6,128)
    const float* n2w     = (const float*)d_in[18];
    const float* n2b     = (const float*)d_in[19];
    const float* f1w     = (const float*)d_in[20];  // (6,512,128)
    const float* f1b     = (const float*)d_in[21];  // (6,512)
    const float* f2w     = (const float*)d_in[22];  // (6,128,512)
    const float* f2b     = (const float*)d_in[23];  // (6,128)
    const float* nfw     = (const float*)d_in[24];
    const float* nfb     = (const float*)d_in[25];
    const float* h1w     = (const float*)d_in[26];
    const float* h1b     = (const float*)d_in[27];
    const float* h2w     = (const float*)d_in[28];
    const float* h2b     = (const float*)d_in[29];
    (void)in_sizes; (void)n_in; (void)out_size; (void)ws_size;

    // ---- float workspace pool ----
    float* ws = (float*)d_ws;
    size_t off = 0;
    float* xpatch  = ws + off; off += (size_t)PROWS * DM;
    float* x       = ws + off; off += (size_t)R_ROWS * DM;
    float* xzb[2], *ub[2], *xdb[2], *dtb[2];
    for (int j = 0; j < 2; ++j) { xzb[j] = ws + off; off += (size_t)R_ROWS * 512; }
    for (int j = 0; j < 2; ++j) { ub[j]  = ws + off; off += (size_t)R_ROWS * DI; }
    for (int j = 0; j < 2; ++j) { xdb[j] = ws + off; off += (size_t)R_ROWS * 40; }
    for (int j = 0; j < 2; ++j) { dtb[j] = ws + off; off += (size_t)R_ROWS * DI; }

    // ---- fp16 workspace pool (starts 32B-aligned: all offsets above are
    //      multiples of 8 floats) ----
    _Float16* hp = (_Float16*)(ws + off);
    size_t ho = 0;
    _Float16* patches_h = hp + ho; ho += (size_t)PROWS * 256;
    _Float16* xn_h      = hp + ho; ho += (size_t)R_ROWS * DM;
    _Float16* uh[2], *yh[2];
    for (int j = 0; j < 2; ++j) { uh[j] = hp + ho; ho += (size_t)R_ROWS * DI; }
    for (int j = 0; j < 2; ++j) { yh[j] = hp + ho; ho += (size_t)R_ROWS * DI; }
    _Float16* xcat_h = hp + ho; ho += (size_t)R_ROWS * DI;
    _Float16* hdd_h  = hp + ho; ho += (size_t)R_ROWS * 512;
    // fp16 weights
    _Float16* wpatch_h = hp + ho; ho += (size_t)DM * 256;
    _Float16* win_h    = hp + ho; ho += (size_t)12 * 512 * DM;
    _Float16* wx_h     = hp + ho; ho += (size_t)12 * 40 * DI;
    _Float16* wout_h   = hp + ho; ho += (size_t)12 * DM * DI;
    _Float16* merge_h  = hp + ho; ho += (size_t)6 * DM * DI;
    _Float16* f1_h     = hp + ho; ho += (size_t)6 * 512 * DM;
    _Float16* f2_h     = hp + ho; ho += (size_t)6 * DM * 512;

    const int TPB = 256;

    // ---- one-time weight conversion to fp16 ----
    launch_cvt(stream, patch_w, wpatch_h, DM * 256);
    launch_cvt(stream, mWin,    win_h,    12 * 512 * DM);
    launch_cvt(stream, mWx,     wx_h,     12 * 40 * DI);
    launch_cvt(stream, mWout,   wout_h,   12 * DM * DI);
    launch_cvt(stream, mergew,  merge_h,  6 * DM * DI);
    launch_cvt(stream, f1w,     f1_h,     6 * 512 * DM);
    launch_cvt(stream, f2w,     f2_h,     6 * DM * 512);

    // ---- patch embedding: im2col (fp16) + WMMA GEMM, then assemble ----
    im2col_kernel<<<(PROWS * 256 + TPB - 1) / TPB, TPB, 0, stream>>>(img, patches_h);
    launch_gemm(stream, patches_h, 256, wpatch_h, 256, nullptr, nullptr, 0,
                xpatch, DM, nullptr, 0, PROWS, DM, 256, 0);
    assemble_x_kernel<<<(R_ROWS * DM + TPB - 1) / TPB, TPB, 0, stream>>>(
        xpatch, patch_b, cls_tok, pos, x);

    for (int i = 0; i < 6; ++i) {
        // --- Mamba mixer ---
        layernorm_kernel<<<R_ROWS, DM, 0, stream>>>(x, n1w + i * DM, n1b + i * DM, xn_h);

        for (int j = 0; j < 2; ++j) {
            const int ij = i * 2 + j;
            const _Float16* Win  = win_h  + (size_t)ij * 512 * DM;
            const _Float16* Wx   = wx_h   + (size_t)ij * 40 * DI;
            const _Float16* Wout = wout_h + (size_t)ij * DM * DI;
            const float* convw = mConvw + (size_t)ij * DI * 4;
            const float* convb = mConvb + (size_t)ij * DI;
            const float* Wdt   = mWdt   + (size_t)ij * DI * DTR;
            const float* bdt   = mbdt   + (size_t)ij * DI;
            const float* Alog  = mAlog  + (size_t)ij * DI * DS;
            const float* Dp    = mDp    + (size_t)ij * DI;

            // in_proj: xn (R x 128) x Win^T -> xz (R x 512, f32)
            launch_gemm(stream, xn_h, DM, Win, DM, nullptr, nullptr, 0,
                        xzb[j], 512, nullptr, 0, R_ROWS, 512, DM, 0);
            // causal depthwise conv + SiLU (direction-aware indexing)
            conv_silu_kernel<<<(R_ROWS * DI + TPB - 1) / TPB, TPB, 0, stream>>>(
                xzb[j], convw, convb, ub[j], uh[j], j);
            // x_proj: u (R x 256) x Wx^T -> xdbl (R x 40, f32)
            launch_gemm(stream, uh[j], DI, Wx, DI, nullptr, nullptr, 0,
                        xdb[j], 40, nullptr, 0, R_ROWS, 40, DI, 0);
            // dt_proj + softplus
            dt_kernel<<<(R_ROWS * DI + TPB - 1) / TPB, TPB, 0, stream>>>(
                xdb[j], 40, Wdt, bdt, dtb[j]);
            // selective scan (+ silu(z) gate), fp16 out
            scan_kernel<<<B_SZ, DI, 0, stream>>>(
                dtb[j], ub[j], xdb[j], 40, xzb[j], Alog, Dp, yh[j], j);
            // out_proj into concat buffer columns [j*128, j*128+128), fp16
            launch_gemm(stream, yh[j], DI, Wout, DI, nullptr, nullptr, 0,
                        nullptr, 0, xcat_h + j * DM, DI, R_ROWS, DM, DI, 0);
        }
        // merge: x = x + xcat @ mergew^T + mergeb  (residual fused)
        launch_gemm(stream, xcat_h, DI, merge_h + (size_t)i * DM * DI, DI,
                    mergeb + i * DM, x, DM, x, DM, nullptr, 0,
                    R_ROWS, DM, DI, 0);

        // --- MLP ---
        layernorm_kernel<<<R_ROWS, DM, 0, stream>>>(x, n2w + i * DM, n2b + i * DM, xn_h);
        launch_gemm(stream, xn_h, DM, f1_h + (size_t)i * 512 * DM, DM,
                    f1b + i * 512, nullptr, 0, nullptr, 0, hdd_h, 512,
                    R_ROWS, 512, DM, 1);
        launch_gemm(stream, hdd_h, 512, f2_h + (size_t)i * DM * 512, 512,
                    f2b + i * DM, x, DM, x, DM, nullptr, 0,
                    R_ROWS, DM, 512, 0);
    }

    head_kernel<<<B_SZ, DM, 0, stream>>>(x, nfw, nfb, h1w, h1b, h2w, h2b, (float*)d_out);
}
